// GNN_6837587935613
// MI455X (gfx1250) — matrix operand order repaired and checked by reading, not compiled, not run on hardware
//
#include <hip/hip_runtime.h>
#include <hip/hip_bf16.h>

#define GDIM  512
#define H1    128
#define RREL  8
#define NRELP 9   // 8 relations + root weight treated as relation 8

typedef __attribute__((ext_vector_type(16))) _Float16 v16h;
typedef __attribute__((ext_vector_type(8)))  _Float16 v8h;
typedef __attribute__((ext_vector_type(4)))  _Float16 v4h;
typedef __attribute__((ext_vector_type(8)))  float    v8f;

// ---------------------------------------------------------------- helpers ----
static __device__ __forceinline__ v16h cat16(v8h lo, v8h hi) {
  return __builtin_shufflevector(lo, hi, 0,1,2,3,4,5,6,7,8,9,10,11,12,13,14,15);
}
// A fragment (16x32 f16, wave32): lane holds row m=lane%16; halfs 0..15 = K[kb..kb+15],
// kb = (lane>=16)?16:0  -> two contiguous b128 loads.
static __device__ __forceinline__ v16h load_afrag(const _Float16* p) {
  v8h lo = *(const v8h*)(p);
  v8h hi = *(const v8h*)(p + 8);
  return cat16(lo, hi);
}
// B fragment: pre-packed so each lane's 16 halfs are contiguous -> two b128 loads.
static __device__ __forceinline__ v16h load_bfrag(const _Float16* p) {
  v8h lo = *(const v8h*)(p);
  v8h hi = *(const v8h*)(p + 8);
  return cat16(lo, hi);
}
// order-preserving float -> uint key (for atomicMax-based segment max)
static __device__ __forceinline__ unsigned fkey(float f) {
  unsigned u = __float_as_uint(f);
  return (u & 0x80000000u) ? ~u : (u | 0x80000000u);
}
static __device__ __forceinline__ float fkeyinv(unsigned kk) {
  unsigned u = (kk & 0x80000000u) ? (kk & 0x7FFFFFFFu) : ~kk;
  return __uint_as_float(u);
}

// ------------------------------------------------------------------ init -----
__global__ void init_ws(float* __restrict__ agg1, float* __restrict__ cnt,
                        float* __restrict__ ssum, unsigned* __restrict__ smk,
                        float* __restrict__ csum, float* __restrict__ csq, int N) {
  int g = blockIdx.x * blockDim.x + threadIdx.x;
  if (g < N * H1)  agg1[g] = 0.f;
  if (g < N * RREL) cnt[g] = 0.f;
  if (g < N) { ssum[g] = 0.f; smk[g] = 0x007FFFFFu; /* fkey(-inf) */ }
  if (g < H1) { csum[g] = 0.f; csq[g] = 0.f; }
}

// ------------------------------------------------- weight packing (f16 B) ----
// layout: (((rel*ksteps + kstep)*8 + ntile)*32 + lane) * 16 halfs
__global__ void pack_big(const float* __restrict__ Wrel, const float* __restrict__ Wroot,
                         _Float16* __restrict__ Wp) {
  int g = blockIdx.x * blockDim.x + threadIdx.x;
  if (g >= NRELP * 16 * 8 * 32) return;
  int lane  = g & 31;
  int ntile = (g >> 5) & 7;
  int kstep = (g >> 8) & 15;
  int rel   = g >> 12;
  const float* W = (rel < RREL) ? (Wrel + (size_t)rel * GDIM * H1) : Wroot;
  int n     = ntile * 16 + (lane & 15);
  int kbase = kstep * 32 + (lane >> 4) * 16;
  _Float16* dst = Wp + (size_t)g * 16;
#pragma unroll
  for (int j = 0; j < 16; ++j)
    dst[j] = (_Float16)W[(size_t)(kbase + j) * H1 + n];
}

__global__ void pack_small(const float* __restrict__ Wq, const float* __restrict__ Wk,
                           const float* __restrict__ Wv, const float* __restrict__ Ws,
                           _Float16* __restrict__ Wsp) {
  int g = blockIdx.x * blockDim.x + threadIdx.x;
  if (g >= 4 * 4 * 8 * 32) return;
  int lane  = g & 31;
  int ntile = (g >> 5) & 7;
  int kstep = (g >> 8) & 3;
  int w     = g >> 10;
  const float* W = (w == 0) ? Wq : (w == 1) ? Wk : (w == 2) ? Wv : Ws;
  int n     = ntile * 16 + (lane & 15);
  int kbase = kstep * 32 + (lane >> 4) * 16;
  _Float16* dst = Wsp + (size_t)g * 16;
#pragma unroll
  for (int j = 0; j < 16; ++j)
    dst[j] = (_Float16)W[(size_t)(kbase + j) * H1 + n];
}

__global__ void convert_x(const float* __restrict__ x, _Float16* __restrict__ xh, int total4) {
  int g = blockIdx.x * blockDim.x + threadIdx.x;
  if (g >= total4) return;
  float4 f = ((const float4*)x)[g];
  v4h h = { (_Float16)f.x, (_Float16)f.y, (_Float16)f.z, (_Float16)f.w };
  *(v4h*)(xh + (size_t)g * 4) = h;
}

// ----------------------------------------------------- RGCN batched GEMM -----
// xW[n, rel, h] = x[n,:] @ W[rel][:,h]   (rel 8 == W_root), f32 acc -> f16 store
__global__ __launch_bounds__(256) void rgcn_gemm(const _Float16* __restrict__ xh,
                                                 const _Float16* __restrict__ Wp,
                                                 _Float16* __restrict__ xW) {
  const int lane = threadIdx.x & 31;
  const int wave = threadIdx.x >> 5;            // ntile 0..7 -> 128 cols
  const int rowtile = blockIdx.x;               // N/16 exact (50000/16=3125)
  const int rel = blockIdx.y;                   // 0..8
  const int row = rowtile * 16 + (lane & 15);
  const int kb  = (lane >> 4) * 16;
  const _Float16* a_base = xh + (size_t)row * GDIM + kb;
  v8f c = {};
#pragma unroll
  for (int ks = 0; ks < 16; ++ks) {
    v16h a = load_afrag(a_base + ks * 32);
    v16h b = load_bfrag(Wp + ((((size_t)rel * 16 + ks) * 8 + wave) * 32 + lane) * 16);
    c = __builtin_amdgcn_wmma_f32_16x16x32_f16(false, a, false, b, (short)0, c, false, false);
  }
  const int n = wave * 16 + (lane & 15);
  const int mbase = (lane >> 4) * 8;
#pragma unroll
  for (int j = 0; j < 8; ++j) {
    int m = rowtile * 16 + mbase + j;
    xW[((size_t)m * NRELP + rel) * H1 + n] = (_Float16)c[j];
  }
}

// --------------------------------------------------------- edge kernels ------
__global__ void count_edges(const int* __restrict__ eidx, const int* __restrict__ etyp,
                            float* __restrict__ cnt, int E) {
  int e = blockIdx.x * blockDim.x + threadIdx.x;
  if (e >= E) return;
  int dst = eidx[E + e];
  atomicAdd(&cnt[(size_t)dst * RREL + etyp[e]], 1.0f);
}

// wave-per-edge: agg1[dst] += xW[src, etype] / denom
__global__ __launch_bounds__(256) void rgcn_agg(const int* __restrict__ eidx,
                                                const int* __restrict__ etyp,
                                                const float* __restrict__ cnt,
                                                const _Float16* __restrict__ xW,
                                                float* __restrict__ agg1, int E) {
  int gw = (blockIdx.x * blockDim.x + threadIdx.x) >> 5;
  int lane = threadIdx.x & 31;
  if (gw >= E) return;
  int src = eidx[gw], dst = eidx[E + gw], et = etyp[gw];
  float inv = 1.0f / fmaxf(cnt[(size_t)dst * RREL + et], 1.0f);
  v4h hv = *(const v4h*)(xW + ((size_t)src * NRELP + et) * H1 + lane * 4);
  float* ag = agg1 + (size_t)dst * H1 + lane * 4;
#pragma unroll
  for (int j = 0; j < 4; ++j) atomicAdd(ag + j, (float)hv[j] * inv);
}

// h = agg1 + x@W_root + b1  ->  f16 for next GEMMs
__global__ void h_finalize(const float* __restrict__ agg1, const _Float16* __restrict__ xW,
                           const float* __restrict__ b1, _Float16* __restrict__ hh, int N) {
  int g = blockIdx.x * blockDim.x + threadIdx.x;
  if (g >= N * H1) return;
  int n = g >> 7, c = g & 127;
  float h = agg1[g] + (float)xW[((size_t)n * NRELP + RREL) * H1 + c] + b1[c];
  hh[g] = (_Float16)h;
}

// ----------------------------------------------- fused Q/K/V/skip GEMMs ------
__global__ __launch_bounds__(256) void qkvs_gemm(const _Float16* __restrict__ hh,
                                                 const _Float16* __restrict__ Wsp,
                                                 const float* __restrict__ bq,
                                                 const float* __restrict__ bk,
                                                 const float* __restrict__ bv,
                                                 const float* __restrict__ bs,
                                                 _Float16* __restrict__ q,
                                                 _Float16* __restrict__ k,
                                                 _Float16* __restrict__ v,
                                                 float* __restrict__ h2) {
  const int lane = threadIdx.x & 31;
  const int wave = threadIdx.x >> 5;
  const int rowtile = blockIdx.x;
  const int y = blockIdx.y;                    // 0=q 1=k 2=v 3=skip
  const int row = rowtile * 16 + (lane & 15);
  const int kb  = (lane >> 4) * 16;
  const _Float16* a_base = hh + (size_t)row * H1 + kb;
  v8f c = {};
#pragma unroll
  for (int ks = 0; ks < 4; ++ks) {
    v16h a = load_afrag(a_base + ks * 32);
    v16h b = load_bfrag(Wsp + ((((size_t)y * 4 + ks) * 8 + wave) * 32 + lane) * 16);
    c = __builtin_amdgcn_wmma_f32_16x16x32_f16(false, a, false, b, (short)0, c, false, false);
  }
  const int n = wave * 16 + (lane & 15);
  const int mbase = (lane >> 4) * 8;
  const float* bias = (y == 0) ? bq : (y == 1) ? bk : (y == 2) ? bv : bs;
  const float bn = bias[n];
  if (y == 3) {
#pragma unroll
    for (int j = 0; j < 8; ++j)
      h2[(size_t)(rowtile * 16 + mbase + j) * H1 + n] = c[j] + bn;
  } else {
    _Float16* o = (y == 0) ? q : (y == 1) ? k : v;
#pragma unroll
    for (int j = 0; j < 8; ++j)
      o[(size_t)(rowtile * 16 + mbase + j) * H1 + n] = (_Float16)(c[j] + bn);
  }
}

// score = dot(q[dst], k[src]) / sqrt(128); segment max via keyed atomicMax
__global__ __launch_bounds__(256) void score_kernel(const int* __restrict__ eidx,
                                                    const _Float16* __restrict__ q,
                                                    const _Float16* __restrict__ k,
                                                    float* __restrict__ score,
                                                    unsigned* __restrict__ smk, int E) {
  int gw = (blockIdx.x * blockDim.x + threadIdx.x) >> 5;
  int lane = threadIdx.x & 31;
  if (gw >= E) return;
  int src = eidx[gw], dst = eidx[E + gw];
  v4h qv = *(const v4h*)(q + (size_t)dst * H1 + lane * 4);
  v4h kv = *(const v4h*)(k + (size_t)src * H1 + lane * 4);
  float s = 0.f;
#pragma unroll
  for (int j = 0; j < 4; ++j) s += (float)qv[j] * (float)kv[j];
#pragma unroll
  for (int m = 16; m >= 1; m >>= 1) s += __shfl_xor(s, m, 32);
  if (lane == 0) {
    s *= 0.08838834764831845f;   // 1/sqrt(128)
    score[gw] = s;
    atomicMax(&smk[dst], fkey(s));
  }
}

__global__ void expsum_kernel(const int* __restrict__ eidx, const float* __restrict__ score,
                              const unsigned* __restrict__ smk, float* __restrict__ ex,
                              float* __restrict__ ssum, int E) {
  int e = blockIdx.x * blockDim.x + threadIdx.x;
  if (e >= E) return;
  int dst = eidx[E + e];
  float m = fkeyinv(smk[dst]);
  if (!isfinite(m)) m = 0.f;
  float x = __expf(score[e] - m);
  ex[e] = x;
  atomicAdd(&ssum[dst], x);
}

__global__ __launch_bounds__(256) void attn_agg(const int* __restrict__ eidx,
                                                const float* __restrict__ ex,
                                                const float* __restrict__ ssum,
                                                const _Float16* __restrict__ v,
                                                float* __restrict__ h2, int E) {
  int gw = (blockIdx.x * blockDim.x + threadIdx.x) >> 5;
  int lane = threadIdx.x & 31;
  if (gw >= E) return;
  int src = eidx[gw], dst = eidx[E + gw];
  float alpha = ex[gw] / fmaxf(ssum[dst], 1e-16f);
  v4h vv = *(const v4h*)(v + (size_t)src * H1 + lane * 4);
  float* o = h2 + (size_t)dst * H1 + lane * 4;
#pragma unroll
  for (int j = 0; j < 4; ++j) atomicAdd(o + j, alpha * (float)vv[j]);
}

// ------------------------------------------------------ batchnorm + lrelu ----
__global__ void bn_stats(const float* __restrict__ h2, float* __restrict__ csum,
                         float* __restrict__ csq, int N) {
  int g = blockIdx.x * blockDim.x + threadIdx.x;
  int c = g & 127;
  int slot = g >> 7;
  int nslots = (gridDim.x * blockDim.x) >> 7;
  float s = 0.f, s2 = 0.f;
  for (int n = slot; n < N; n += nslots) {
    float x = h2[(size_t)n * H1 + c];
    s += x; s2 += x * x;
  }
  atomicAdd(&csum[c], s);
  atomicAdd(&csq[c], s2);
}

__global__ void bn_apply(const float* __restrict__ h2, const float* __restrict__ csum,
                         const float* __restrict__ csq, const float* __restrict__ gamma,
                         const float* __restrict__ beta, float* __restrict__ out, int N) {
  int g = blockIdx.x * blockDim.x + threadIdx.x;
  if (g >= N * H1) return;
  int c = g & 127;
  float invN = 1.0f / (float)N;
  float mu  = csum[c] * invN;
  float var = csq[c] * invN - mu * mu;
  float xn  = (h2[g] - mu) * rsqrtf(var + 1e-5f);
  float o   = gamma[c] * xn + beta[c];
  out[g] = (o >= 0.f) ? o : 0.01f * o;
}

// ------------------------------------------------------------- launcher ------
extern "C" void kernel_launch(void* const* d_in, const int* in_sizes, int n_in,
                              void* d_out, int out_size, void* d_ws, size_t ws_size,
                              hipStream_t stream) {
  const float* x      = (const float*)d_in[0];
  const int*   eidx   = (const int*)  d_in[1];
  const int*   etyp   = (const int*)  d_in[2];
  const float* W_rel  = (const float*)d_in[3];
  const float* W_root = (const float*)d_in[4];
  const float* b1     = (const float*)d_in[5];
  const float* Wq     = (const float*)d_in[6];
  const float* bq     = (const float*)d_in[7];
  const float* Wk     = (const float*)d_in[8];
  const float* bk     = (const float*)d_in[9];
  const float* Wv     = (const float*)d_in[10];
  const float* bv     = (const float*)d_in[11];
  const float* Wsk    = (const float*)d_in[12];
  const float* bsk    = (const float*)d_in[13];
  const float* gamma  = (const float*)d_in[14];
  const float* beta   = (const float*)d_in[15];
  float* out = (float*)d_out;
  const int N = in_sizes[0] / GDIM;   // 50000 (multiple of 16)
  const int E = in_sizes[2];          // 800000
  (void)n_in; (void)out_size; (void)ws_size;

  size_t off = 0;
  auto alloc = [&](size_t bytes) -> void* {
    off = (off + 255) & ~(size_t)255;
    void* p = (char*)d_ws + off;
    off += bytes;
    return p;
  };
  _Float16* xh   = (_Float16*)alloc((size_t)N * GDIM * 2);
  _Float16* Wp   = (_Float16*)alloc((size_t)NRELP * GDIM * H1 * 2);
  _Float16* Wsp  = (_Float16*)alloc((size_t)4 * H1 * H1 * 2);
  _Float16* xW   = (_Float16*)alloc((size_t)N * NRELP * H1 * 2);
  float*    cnt  = (float*)   alloc((size_t)N * RREL * 4);
  float*    agg1 = (float*)   alloc((size_t)N * H1 * 4);
  _Float16* hh   = (_Float16*)alloc((size_t)N * H1 * 2);
  _Float16* q    = (_Float16*)alloc((size_t)N * H1 * 2);
  _Float16* k    = (_Float16*)alloc((size_t)N * H1 * 2);
  _Float16* v    = (_Float16*)alloc((size_t)N * H1 * 2);
  float*    h2   = (float*)   alloc((size_t)N * H1 * 4);
  float*    score= (float*)   alloc((size_t)E * 4);
  float*    ex   = (float*)   alloc((size_t)E * 4);
  unsigned* smk  = (unsigned*)alloc((size_t)N * 4);
  float*    ssum = (float*)   alloc((size_t)N * 4);
  float*    csum = (float*)   alloc(H1 * 4);
  float*    csq  = (float*)   alloc(H1 * 4);

  init_ws   <<<(N * H1 + 255) / 256, 256, 0, stream>>>(agg1, cnt, ssum, smk, csum, csq, N);
  pack_big  <<<(NRELP * 16 * 8 * 32 + 255) / 256, 256, 0, stream>>>(W_rel, W_root, Wp);
  pack_small<<<(4 * 4 * 8 * 32 + 255) / 256, 256, 0, stream>>>(Wq, Wk, Wv, Wsk, Wsp);
  convert_x <<<((N * GDIM / 4) + 255) / 256, 256, 0, stream>>>(x, xh, N * GDIM / 4);
  count_edges<<<(E + 255) / 256, 256, 0, stream>>>(eidx, etyp, cnt, E);

  rgcn_gemm <<<dim3(N / 16, NRELP), 256, 0, stream>>>(xh, Wp, xW);
  rgcn_agg  <<<(E * 32 + 255) / 256, 256, 0, stream>>>(eidx, etyp, cnt, xW, agg1, E);
  h_finalize<<<(N * H1 + 255) / 256, 256, 0, stream>>>(agg1, xW, b1, hh, N);

  qkvs_gemm <<<dim3(N / 16, 4), 256, 0, stream>>>(hh, Wsp, bq, bk, bv, bsk, q, k, v, h2);
  score_kernel<<<(E * 32 + 255) / 256, 256, 0, stream>>>(eidx, q, k, score, smk, E);
  expsum_kernel<<<(E + 255) / 256, 256, 0, stream>>>(eidx, score, smk, ex, ssum, E);
  attn_agg  <<<(E * 32 + 255) / 256, 256, 0, stream>>>(eidx, ex, ssum, v, h2, E);

  bn_stats  <<<1024, 256, 0, stream>>>(h2, csum, csq, N);
  bn_apply  <<<(N * H1 + 255) / 256, 256, 0, stream>>>(h2, csum, csq, gamma, beta, out, N);
}
